// Model_17540646437378
// MI455X (gfx1250) — matrix operand order repaired and checked by reading, not compiled, not run on hardware
//
#include <hip/hip_runtime.h>
#include <hip/hip_bf16.h>

typedef _Float16 f16;
typedef __attribute__((ext_vector_type(16))) _Float16 v16h;
typedef __attribute__((ext_vector_type(8)))  _Float16 v8h;
typedef __attribute__((ext_vector_type(8)))  float    v8f;

#define RNN_HID   64
#define INPUT_DIM 35
#define SEQ_LEN   48
#define BATCH     4096
#define KDIM      160   // padded concat: [h(0..63) | x_c(64..98) | m(99..133) | pad(134..159)]
#define NG        256   // 4*H gate width

// ws layout (halves / floats)
#define WCP_HALVES (5*16*512)   // gates weights, 5 K-chunks x 16 N-tiles x (32 lanes x 16 halves)
#define WRP_HALVES (2*3*512)    // W_reg,        2 K-chunks x  3 N-tiles

// ---------------------------------------------------------------------------
// Pack weights into f16 WMMA B-fragment layout.
// B 32x16 fragment, contiguous k order (same as A here): lane L holds column N = L%16; half h holds
// K = h + 16*(L/16) within the 32-deep K-chunk.
// Combined gates weight Wc[K][N]: K<64 -> W_hh[N][K]; 64..98 -> W_ih[N][K-64]
// (x_c part); 99..133 -> W_ih[N][35+K-99] (mask part); else 0.
// ---------------------------------------------------------------------------
__global__ void pack_weights(const float* __restrict__ W_ih, const float* __restrict__ W_hh,
                             const float* __restrict__ b_ih, const float* __restrict__ b_hh,
                             const float* __restrict__ W_reg,
                             f16* __restrict__ wcp, f16* __restrict__ wrp,
                             float* __restrict__ bg) {
  int tid = threadIdx.x;
  for (int i = tid; i < WCP_HALVES; i += 256) {
    int c    = i >> 13;          // /8192  K-chunk
    int r    = i & 8191;
    int nt   = r >> 9;           // /512   N-tile
    int q    = r & 511;
    int lane = q >> 4;
    int h    = q & 15;
    int n    = nt * 16 + (lane & 15);
    int K    = c * 32 + h + 16 * (lane >> 4);
    float v;
    if (K < 64)       v = W_hh[n * 64 + K];
    else if (K < 99)  v = W_ih[n * 70 + (K - 64)];
    else if (K < 134) v = W_ih[n * 70 + 35 + (K - 99)];
    else              v = 0.f;
    wcp[i] = (f16)v;
  }
  for (int i = tid; i < WRP_HALVES; i += 256) {
    int c    = i / 1536;
    int r    = i % 1536;
    int nt   = r >> 9;
    int q    = r & 511;
    int lane = q >> 4;
    int h    = q & 15;
    int n    = nt * 16 + (lane & 15);
    int K    = c * 32 + h + 16 * (lane >> 4);   // K < 64
    wrp[i] = (f16)((n < INPUT_DIM) ? W_reg[n * RNN_HID + K] : 0.f);
  }
  for (int i = tid; i < NG; i += 256) bg[i] = b_ih[i] + b_hh[i];
}

// Per-step reciprocal mask sums: rdenom[t] = 1 / (sum_{b,d} masks[b,t,d] + 1e-5)
__global__ void mask_denom(const float* __restrict__ masks, float* __restrict__ rdenom) {
  int t = blockIdx.x;
  __shared__ float red[256];
  float s = 0.f;
  for (int e = threadIdx.x; e < BATCH * INPUT_DIM; e += 256) {
    int b = e / INPUT_DIM, d = e % INPUT_DIM;
    s += masks[(size_t)b * SEQ_LEN * INPUT_DIM + (size_t)t * INPUT_DIM + d];
  }
  red[threadIdx.x] = s; __syncthreads();
  for (int off = 128; off > 0; off >>= 1) {
    if (threadIdx.x < off) red[threadIdx.x] += red[threadIdx.x + off];
    __syncthreads();
  }
  if (threadIdx.x == 0) rdenom[t] = 1.f / (red[0] + 1e-5f);
}

__global__ void init_out(float* out) { if (threadIdx.x == 0) out[0] = 0.f; }

__device__ __forceinline__ float sigm(float x) { return 1.f / (1.f + __expf(-x)); }
__device__ __forceinline__ float tanh_fast(float x) {
  float e = __expf(2.f * x);
  return (e - 1.f) / (e + 1.f);
}

// Workgroup barrier that drains only LDS traffic (DScnt). Global stores
// (imputations, write-only & disjoint) keep draining in the background;
// global loads are synchronized by register data deps. Avoids the
// s_wait_storecnt/loadcnt stalls __syncthreads() would insert 144x.
__device__ __forceinline__ void block_sync_lds() {
  asm volatile("s_wait_dscnt 0x0\n\t"
               "s_barrier_signal -1\n\t"
               "s_barrier_wait -1" ::: "memory");
}

// ---------------------------------------------------------------------------
// Main fused RITS kernel: one block = 16 batch rows, 48 timesteps resident.
// 3 barriers per step: [xhat WMMA + x_c from C-fragment] | [gates GEMM] | [LSTM update]
// ---------------------------------------------------------------------------
__global__ void __launch_bounds__(256) rits_main(
    const float* __restrict__ x_t, const float* __restrict__ masks,
    const f16* __restrict__ wcp, const f16* __restrict__ wrp,
    const float* __restrict__ bg, const float* __restrict__ b_reg,
    const float* __restrict__ W_out, const float* __restrict__ b_out,
    const float* __restrict__ rdenomG, float* __restrict__ out) {

  __shared__ __align__(16) f16 As[16 * KDIM];     // A-operand staging (h | x_c | m | pad)
  __shared__ float gBuf[16 * NG];
  __shared__ float cS[16 * RNN_HID];
  __shared__ float hF[16 * RNN_HID];
  __shared__ float dnm[SEQ_LEN];                  // reciprocals
  __shared__ float red[256];

  const int tid  = threadIdx.x;
  const int lane = tid & 31;          // wave32 (gfx1250)
  const int wave = tid >> 5;
  const int hi   = lane >> 4;
  const int l16  = lane & 15;
  const int bbase = blockIdx.x * 16;

  // ---- init resident state ----
  for (int i = tid; i < 16 * KDIM; i += 256) As[i] = (f16)0.f;     // h=0, pad region=0
  for (int i = tid; i < 16 * RNN_HID; i += 256) cS[i] = 0.f;
  if (tid < SEQ_LEN) dnm[tid] = rdenomG[tid];

  // ---- load B fragments into registers (kept across all 48 steps) ----
  v16h wg[10];
#pragma unroll
  for (int c = 0; c < 5; c++)
#pragma unroll
    for (int s = 0; s < 2; s++) {
      int nt = 2 * wave + s;
      wg[c * 2 + s] = *(const v16h*)(wcp + (size_t)((c * 16 + nt) * 32 + lane) * 16);
    }
  const int wvr = (wave < 3) ? wave : 0;
  v16h wr0 = *(const v16h*)(wrp + (size_t)((0 * 3 + wvr) * 32 + lane) * 16);
  v16h wr1 = *(const v16h*)(wrp + (size_t)((1 * 3 + wvr) * 32 + lane) * 16);

  const int dcol = wave * 16 + l16;        // regression column owned by this lane (waves 0-2)
  float biasR = 0.f;
  if (wave < 3 && dcol < INPUT_DIM) biasR = b_reg[dcol];
  float biasG0 = bg[(2 * wave)     * 16 + l16];
  float biasG1 = bg[(2 * wave + 1) * 16 + l16];

  __syncthreads();

  // A-fragment gather from LDS staging (contiguous k order, same as packed B):
  // lane L -> row L%16; halves 0..15 = K kbase+16*hi .. +15.
  auto loadA = [&](int kbase) -> v16h {
    const f16* p = As + l16 * KDIM + kbase + 16 * hi;
    v8h lo = *(const v8h*)p;
    v8h up = *(const v8h*)(p + 8);
    v16h a;
#pragma unroll
    for (int i = 0; i < 8; i++) { a[i] = lo[i]; a[8 + i] = up[i]; }
    return a;
  };

  float lossAcc = 0.f;

  for (int t = 0; t < SEQ_LEN; t++) {
    // ================= phase 1 (waves 0-2): x_hat WMMA + x_c directly
    // from the C fragment (lane = column d, VGPR j = row j+8*hi).
    if (wave < 3) {
      const bool valid = (dcol < INPUT_DIM);
      // prefetch x, m for this lane's column (hidden behind the WMMAs)
      float xv[8], mv[8];
#pragma unroll
      for (int j = 0; j < 8; j++) { xv[j] = 0.f; mv[j] = 0.f; }
      if (valid) {
#pragma unroll
        for (int j = 0; j < 8; j++) {
          int row = j + 8 * hi;
          size_t idx = ((size_t)(bbase + row) * SEQ_LEN + t) * INPUT_DIM + dcol;
          xv[j] = x_t[idx];
          mv[j] = masks[idx];
        }
      }
      v16h a0 = loadA(0);
      v16h a1 = loadA(32);
      v8f acc;
#pragma unroll
      for (int j = 0; j < 8; j++) acc[j] = biasR;
      acc = __builtin_amdgcn_wmma_f32_16x16x32_f16(false, a0, false, wr0, (short)0, acc, false, false);
      acc = __builtin_amdgcn_wmma_f32_16x16x32_f16(false, a1, false, wr1, (short)0, acc, false, false);

      float num = 0.f;
      if (valid) {
#pragma unroll
        for (int j = 0; j < 8; j++) {
          int row = j + 8 * hi;
          float xh = acc[j];
          float xc = mv[j] * xv[j] + (1.f - mv[j]) * xh;
          num += fabsf(xv[j] - xh) * mv[j];
          As[row * KDIM + 64 + dcol] = (f16)xc;
          As[row * KDIM + 99 + dcol] = (f16)mv[j];
          out[1 + BATCH + ((size_t)(bbase + row) * SEQ_LEN + t) * INPUT_DIM + dcol] = xc;
        }
      }
      lossAcc += num * dnm[t];
    }
    block_sync_lds();

    // ================= phase 2: gates = [h|x_c|m] @ Wc + b  (M=16 N=256 K=160)
    v16h a[5];
#pragma unroll
    for (int c = 0; c < 5; c++) a[c] = loadA(32 * c);
    v8f ac0, ac1;
#pragma unroll
    for (int j = 0; j < 8; j++) { ac0[j] = biasG0; ac1[j] = biasG1; }
#pragma unroll
    for (int c = 0; c < 5; c++) {
      ac0 = __builtin_amdgcn_wmma_f32_16x16x32_f16(false, a[c], false, wg[2 * c],     (short)0, ac0, false, false);
      ac1 = __builtin_amdgcn_wmma_f32_16x16x32_f16(false, a[c], false, wg[2 * c + 1], (short)0, ac1, false, false);
    }
#pragma unroll
    for (int j = 0; j < 8; j++) {
      int row = j + 8 * hi;
      gBuf[row * NG + (2 * wave)     * 16 + l16] = ac0[j];
      gBuf[row * NG + (2 * wave + 1) * 16 + l16] = ac1[j];
    }
    block_sync_lds();

    // ================= phase 3: LSTM elementwise update (1024 items / 256 thr)
#pragma unroll
    for (int it = 0; it < 4; it++) {
      int idx = tid + 256 * it;
      int r = idx >> 6, k = idx & 63;
      float i_ = sigm(gBuf[r * NG + k]);
      float f_ = sigm(gBuf[r * NG + 64 + k]);
      float g_ = tanh_fast(gBuf[r * NG + 128 + k]);
      float o_ = sigm(gBuf[r * NG + 192 + k]);
      float c_ = f_ * cS[r * RNN_HID + k] + i_ * g_;
      float h_ = o_ * tanh_fast(c_);
      cS[r * RNN_HID + k] = c_;
      hF[r * RNN_HID + k] = h_;
      As[r * KDIM + k] = (f16)h_;       // feeds next step's WMMAs directly
    }
    block_sync_lds();
  }

  // ---- y_h = sigmoid(h @ W_out.T + b_out) ----
  if (tid < 16) {
    float s = b_out[0];
#pragma unroll 8
    for (int k = 0; k < RNN_HID; k++) s += hF[tid * RNN_HID + k] * W_out[k];
    out[1 + bbase + tid] = sigm(s);
  }

  // ---- block loss reduction -> atomic ----
  red[tid] = lossAcc; __syncthreads();
  for (int off = 128; off > 0; off >>= 1) {
    if (tid < off) red[tid] += red[tid + off];
    __syncthreads();
  }
  if (tid == 0) atomicAdd(out, red[0] * (1.f / SEQ_LEN));
}

// ---------------------------------------------------------------------------
extern "C" void kernel_launch(void* const* d_in, const int* in_sizes, int n_in,
                              void* d_out, int out_size, void* d_ws, size_t ws_size,
                              hipStream_t stream) {
  const float* x_t   = (const float*)d_in[0];
  const float* masks = (const float*)d_in[1];
  // d_in[2]=deltas, [3]=evals, [4]=eval_masks, [5]=is_train : unused (dead code / y_loss==0)
  const float* W_ih  = (const float*)d_in[6];
  const float* W_hh  = (const float*)d_in[7];
  const float* b_ih  = (const float*)d_in[8];
  const float* b_hh  = (const float*)d_in[9];
  const float* W_reg = (const float*)d_in[10];
  const float* b_reg = (const float*)d_in[11];
  const float* W_out = (const float*)d_in[14];
  const float* b_out = (const float*)d_in[15];
  float* out = (float*)d_out;

  f16*   wcp   = (f16*)d_ws;
  f16*   wrp   = wcp + WCP_HALVES;
  float* bg    = (float*)(wrp + WRP_HALVES);
  float* rden  = bg + NG;

  init_out<<<1, 32, 0, stream>>>(out);
  pack_weights<<<1, 256, 0, stream>>>(W_ih, W_hh, b_ih, b_hh, W_reg, wcp, wrp, bg);
  mask_denom<<<SEQ_LEN, 256, 0, stream>>>(masks, rden);
  rits_main<<<BATCH / 16, 256, 0, stream>>>(x_t, masks, wcp, wrp, bg, b_reg,
                                            W_out, b_out, rden, out);
}
